// LatticeRNN_23313082482921
// MI455X (gfx1250) — matrix operand order repaired and checked by reading, not compiled, not run on hardware
//
#include <hip/hip_runtime.h>
#include <hip/hip_bf16.h>
#include <stdint.h>

#define B_DIM  8192
#define H_DIM  512
#define GHH    4
#define GWW    2
#define NCELL  8
#define FC_HID 256
#define BN_EPS 1e-5f

typedef __bf16 bh16;
typedef __attribute__((ext_vector_type(16))) __bf16 v16bf;
typedef __attribute__((ext_vector_type(8)))  __bf16 v8bf;
typedef __attribute__((ext_vector_type(8)))  float  v8f;
typedef __attribute__((ext_vector_type(4)))  int    v4i;

#if defined(__has_builtin)
#  if __has_builtin(__builtin_amdgcn_global_load_async_to_lds_b128)
#    define HAVE_ASYNC_LDS 1
#  endif
#  if __has_builtin(__builtin_amdgcn_s_wait_asynccnt)
#    define HAVE_WAIT_ASYNC 1
#  endif
#endif

static __device__ __forceinline__ bh16 f2bf(float f) {
  union { float f; uint32_t u; } v; v.f = f;
  uint32_t r = v.u + 0x7FFFu + ((v.u >> 16) & 1u);   // round-to-nearest-even
  uint16_t h = (uint16_t)(r >> 16);
  return __builtin_bit_cast(bh16, h);
}
static __device__ __forceinline__ float sigm(float x) { return 1.0f / (1.0f + __expf(-x)); }

// 16-byte global -> LDS copy; async (no VGPR round-trip, ASYNCcnt) when available
static __device__ __forceinline__ void copy16_g2l(bh16* lds_dst, const bh16* gsrc) {
#if defined(HAVE_ASYNC_LDS)
  typedef __attribute__((address_space(1))) v4i* gv4ip;   // global int4*
  typedef __attribute__((address_space(3))) v4i* lv4ip;   // LDS int4*
  __builtin_amdgcn_global_load_async_to_lds_b128((gv4ip)gsrc, (lv4ip)lds_dst, 0, 0);
#else
  *reinterpret_cast<v8bf*>(lds_dst) = *reinterpret_cast<const v8bf*>(gsrc);
#endif
}

template <int N> static __device__ __forceinline__ void wait_async_le() {
#if defined(HAVE_ASYNC_LDS)
#  if defined(HAVE_WAIT_ASYNC)
  __builtin_amdgcn_s_wait_asynccnt(N);
#  else
  asm volatile("s_wait_asynccnt %0" :: "n"(N) : "memory");
#  endif
#endif
}

// ---------------- fp32 -> bf16 conversion ----------------
__global__ void cvt_f32_bf16(const float* __restrict__ src, bh16* __restrict__ dst, int n) {
  for (int i = blockIdx.x * blockDim.x + threadIdx.x; i < n; i += gridDim.x * blockDim.x)
    dst[i] = f2bf(src[i]);
}

// ---------------- build [h_left | h_up | h_prev] bf16 buffers ----------------
__global__ void prep_cell(const float* __restrict__ hL, const float* __restrict__ hU,
                          const float* __restrict__ hP,
                          const float* __restrict__ cL, const float* __restrict__ cU,
                          const float* __restrict__ cP,
                          bh16* __restrict__ combH, bh16* __restrict__ combC) {
  const int n = B_DIM * H_DIM;
  for (int i = blockIdx.x * blockDim.x + threadIdx.x; i < n; i += gridDim.x * blockDim.x) {
    int b = i >> 9;
    int h = i & (H_DIM - 1);
    size_t base = (size_t)b * (3 * H_DIM) + h;
    combH[base]             = f2bf(hL ? hL[i] : 0.0f);
    combH[base + H_DIM]     = f2bf(hU ? hU[i] : 0.0f);
    combH[base + 2 * H_DIM] = f2bf(hP[i]);
    combC[base]             = f2bf(cL ? cL[i] : 0.0f);
    combC[base + H_DIM]     = f2bf(cU ? cU[i] : 0.0f);
    combC[base + 2 * H_DIM] = f2bf(cP[i]);
  }
}

// ---------------- bf16 WMMA GEMM: C[M,N] = A[M,K] * W[N,K]^T + epilogue ----------------
// MODE 0: C_bf16 = .. + bias0          (ph)
// MODE 1: C_f32  = .. + bias0          (pc)
// MODE 2: C_f32  = .. + bias0 + bias1 + x[row*8+xoff]*wih[col]   (gates)
// MODE 3: C_f32  = relu(.. + bias0)    (fc1)
#define BLK_K      64
#define LDS_STRIDE 72   // 64 + 8 bf16 pad; 144B row -> 16B-aligned chunk reads

template <int MODE>
__global__ __launch_bounds__(256) void gemm_bf16(
    const bh16* __restrict__ A, const bh16* __restrict__ W,
    const float* __restrict__ bias0, const float* __restrict__ bias1,
    const float* __restrict__ xsrc, const float* __restrict__ wih, int xoff,
    int K, int N, float* __restrict__ Cf, bh16* __restrict__ Cb) {
  __shared__ bh16 sA[2][128 * LDS_STRIDE];
  __shared__ bh16 sB[2][128 * LDS_STRIDE];

  const int tid  = threadIdx.x;
  const int lane = tid & 31;
  const int wave = tid >> 5;
  const int wm   = wave & 3;   // 4 waves along M (32 rows each)
  const int wn   = wave >> 2;  // 2 waves along N (64 cols each)
  const int blockN = blockIdx.x * 128;
  const int blockM = blockIdx.y * 128;
  const int mrow = lane & 15;
  const int grp  = lane >> 4;

  v8f acc[2][4] = {};

  const int NT = K / BLK_K;

  // stage one 128x64 bf16 slab of A and W into LDS buffer `buf`
  auto stage = [&](int buf, int k0) {
#pragma unroll
    for (int it = 0; it < 4; ++it) {
      int c  = tid + it * 256;     // 0..1023 chunks of 8 bf16
      int r  = c >> 3;             // row 0..127
      int cc = (c & 7) * 8;        // col chunk
      copy16_g2l(&sA[buf][r * LDS_STRIDE + cc],
                 A + (size_t)(blockM + r) * K + k0 + cc);
      copy16_g2l(&sB[buf][r * LDS_STRIDE + cc],
                 W + (size_t)(blockN + r) * K + k0 + cc);
    }
  };

  stage(0, 0);
  for (int kt = 0; kt < NT; ++kt) {
    const int cur = kt & 1;
    if (kt + 1 < NT) {
      stage(cur ^ 1, (kt + 1) * BLK_K);   // prefetch next slab into other buffer
      if (kt + 2 < NT) {                   // GL2 prefetch one more slab ahead
        __builtin_prefetch(A + (size_t)(blockM + (tid >> 1)) * K + (kt + 2) * BLK_K, 0, 2);
        __builtin_prefetch(W + (size_t)(blockN + (tid >> 1)) * K + (kt + 2) * BLK_K, 0, 2);
      }
      wait_async_le<8>();                 // retire current slab's 8 async ops
    } else {
      wait_async_le<0>();
    }
    __syncthreads();

#pragma unroll
    for (int ks = 0; ks < 2; ++ks) {
      const int kc = ks * 32;
      // A fragments: lane m = lane&15; lanes 0-15 hold K 0-7 & 16-23, lanes 16-31 K 8-15 & 24-31
      v16bf aF[2];
#pragma unroll
      for (int tm = 0; tm < 2; ++tm) {
        const bh16* base = &sA[cur][(wm * 32 + tm * 16 + mrow) * LDS_STRIDE] + kc;
        v8bf lo = *reinterpret_cast<const v8bf*>(base + grp * 8);
        v8bf hi = *reinterpret_cast<const v8bf*>(base + 16 + grp * 8);
#pragma unroll
        for (int e = 0; e < 8; ++e) { aF[tm][e] = lo[e]; aF[tm][8 + e] = hi[e]; }
      }
      // B fragments: lane n = lane&15; lanes 0-15 hold K 0-7 & 16-23, lanes 16-31 K 8-15 & 24-31
      v16bf bF[4];
#pragma unroll
      for (int tn = 0; tn < 4; ++tn) {
        const bh16* base = &sB[cur][(wn * 64 + tn * 16 + mrow) * LDS_STRIDE] + kc;
        v8bf lo = *reinterpret_cast<const v8bf*>(base + grp * 8);
        v8bf hi = *reinterpret_cast<const v8bf*>(base + 16 + grp * 8);
#pragma unroll
        for (int e = 0; e < 8; ++e) { bF[tn][e] = lo[e]; bF[tn][8 + e] = hi[e]; }
      }
#pragma unroll
      for (int tm = 0; tm < 2; ++tm)
#pragma unroll
        for (int tn = 0; tn < 4; ++tn)
          acc[tm][tn] = __builtin_amdgcn_wmma_f32_16x16x32_bf16(
              false, aF[tm], false, bF[tn], (short)0, acc[tm][tn], false, false);
    }
    __syncthreads();   // protect buffer `cur` from being overwritten next iteration
  }

  // epilogue: D layout — VGPR v: lanes 0-15 row=v, lanes 16-31 row=v+8; col = lane&15
#pragma unroll
  for (int tm = 0; tm < 2; ++tm) {
#pragma unroll
    for (int tn = 0; tn < 4; ++tn) {
      int colg = blockN + wn * 64 + tn * 16 + mrow;
      float bz = bias0[colg];
      float wv = 0.0f;
      if constexpr (MODE == 2) { bz += bias1[colg]; wv = wih[colg]; }
#pragma unroll
      for (int v = 0; v < 8; ++v) {
        int rowg = blockM + wm * 32 + tm * 16 + grp * 8 + v;
        float val = acc[tm][tn][v] + bz;
        if constexpr (MODE == 2) val += xsrc[(size_t)rowg * (GHH * GWW) + xoff] * wv;
        if constexpr (MODE == 3) val = fmaxf(val, 0.0f);
        size_t ci = (size_t)rowg * N + colg;
        if constexpr (MODE == 0) Cb[ci] = f2bf(val);
        else                     Cf[ci] = val;
      }
    }
  }
}

// ---------------- LSTM cell elementwise update ----------------
__global__ void cell_update(const float* __restrict__ gates, const float* __restrict__ pc,
                            float* __restrict__ h_new, float* __restrict__ c_new) {
  const int n = B_DIM * H_DIM;
  for (int i = blockIdx.x * blockDim.x + threadIdx.x; i < n; i += gridDim.x * blockDim.x) {
    int b = i >> 9;
    int h = i & (H_DIM - 1);
    size_t gb = (size_t)b * (4 * H_DIM);
    float ig = gates[gb + h];
    float fg = gates[gb + H_DIM + h];
    float gg = gates[gb + 2 * H_DIM + h];
    float og = gates[gb + 3 * H_DIM + h];
    float cn = sigm(fg) * pc[i] + sigm(ig) * tanhf(gg);
    float hn = sigm(og) * tanhf(cn);
    c_new[i] = cn;
    h_new[i] = hn;
  }
}

// ---------------- final_h/final_c + bf16 concat for FC ----------------
__global__ void final_prep(const float* __restrict__ hlast, const float* __restrict__ clast,
                           const float* __restrict__ h_ext, const float* __restrict__ c_ext,
                           float* __restrict__ final_h, float* __restrict__ final_c,
                           bh16* __restrict__ finB) {
  const int n = B_DIM * H_DIM;
  for (int i = blockIdx.x * blockDim.x + threadIdx.x; i < n; i += gridDim.x * blockDim.x) {
    float fh = hlast[i] + h_ext[i];
    float fc = clast[i] + c_ext[i];
    final_h[i] = fh;
    final_c[i] = fc;
    int b = i >> 9;
    int h = i & (H_DIM - 1);
    finB[(size_t)b * (2 * H_DIM) + h]         = f2bf(fh);
    finB[(size_t)b * (2 * H_DIM) + H_DIM + h] = f2bf(fc);
  }
}

// ---------------- out = hid . W2 + b2 (one wave32 per row) ----------------
__global__ __launch_bounds__(256) void fc2_dot(const float* __restrict__ hid,
                                               const float* __restrict__ W2,
                                               const float* __restrict__ b2,
                                               float* __restrict__ outraw) {
  int wave = threadIdx.x >> 5, lane = threadIdx.x & 31;
  int row = blockIdx.x * 8 + wave;
  const float* hp = hid + (size_t)row * FC_HID + lane * 8;
  const float* wp = W2 + lane * 8;
  float s = 0.0f;
#pragma unroll
  for (int e = 0; e < 8; ++e) s += hp[e] * wp[e];
#pragma unroll
  for (int off = 16; off > 0; off >>= 1) s += __shfl_xor(s, off, 32);
  if (lane == 0) outraw[row] = s + b2[0];
}

// ---------------- batchnorm (training stats) + sigmoid, single block ----------------
__global__ __launch_bounds__(1024) void bn_sigmoid(const float* __restrict__ outraw,
                                                   const float* __restrict__ gamma,
                                                   const float* __restrict__ beta,
                                                   float* __restrict__ out) {
  __shared__ float ssum[1024];
  __shared__ float ssq[1024];
  int t = threadIdx.x;
  float v[8], s = 0.0f, q = 0.0f;
#pragma unroll
  for (int e = 0; e < 8; ++e) { v[e] = outraw[t + e * 1024]; s += v[e]; q += v[e] * v[e]; }
  ssum[t] = s; ssq[t] = q;
  __syncthreads();
  for (int off = 512; off > 0; off >>= 1) {
    if (t < off) { ssum[t] += ssum[t + off]; ssq[t] += ssq[t + off]; }
    __syncthreads();
  }
  float mu  = ssum[0] / (float)B_DIM;
  float var = ssq[0] / (float)B_DIM - mu * mu;
  float inv = rsqrtf(var + BN_EPS);
  float g = gamma[0], be = beta[0];
#pragma unroll
  for (int e = 0; e < 8; ++e) out[t + e * 1024] = sigm(g * (v[e] - mu) * inv + be);
}

extern "C" void kernel_launch(void* const* d_in, const int* in_sizes, int n_in,
                              void* d_out, int out_size, void* d_ws, size_t ws_size,
                              hipStream_t stream) {
  (void)in_sizes; (void)n_in; (void)out_size; (void)ws_size;
  const float* x      = (const float*)d_in[0];
  const float* h_ext  = (const float*)d_in[1];
  const float* c_ext  = (const float*)d_in[2];
  const float* grid_h = (const float*)d_in[3];
  const float* grid_c = (const float*)d_in[4];
  const float* Wh     = (const float*)d_in[5];
  const float* bh     = (const float*)d_in[6];
  const float* Wc     = (const float*)d_in[7];
  const float* bc     = (const float*)d_in[8];
  const float* W_ih   = (const float*)d_in[9];
  const float* W_hh   = (const float*)d_in[10];
  const float* b_ih   = (const float*)d_in[11];
  const float* b_hh   = (const float*)d_in[12];
  const float* W1     = (const float*)d_in[13];
  const float* b1     = (const float*)d_in[14];
  const float* W2     = (const float*)d_in[15];
  const float* b2     = (const float*)d_in[16];
  const float* gamma  = (const float*)d_in[17];
  const float* beta   = (const float*)d_in[18];

  const size_t slice = (size_t)B_DIM * H_DIM;
  float* out     = (float*)d_out;
  float* final_h = out + B_DIM;
  float* final_c = final_h + slice;
  float* new_gh  = final_c + slice;
  float* new_gc  = new_gh + (size_t)NCELL * slice;

  // workspace layout (256B aligned chunks)
  char* wp = (char*)d_ws;
  auto alloc = [&](size_t bytes) -> char* {
    char* p = wp;
    wp += (bytes + 255) & ~(size_t)255;
    return p;
  };
  bh16*  wWh    = (bh16*)alloc((size_t)NCELL * H_DIM * 3 * H_DIM * 2);
  bh16*  wWc    = (bh16*)alloc((size_t)NCELL * H_DIM * 3 * H_DIM * 2);
  bh16*  wWhh   = (bh16*)alloc((size_t)NCELL * 4 * H_DIM * H_DIM * 2);
  bh16*  wW1    = (bh16*)alloc((size_t)FC_HID * 2 * H_DIM * 2);
  bh16*  combH  = (bh16*)alloc((size_t)B_DIM * 3 * H_DIM * 2);
  bh16*  combC  = (bh16*)alloc((size_t)B_DIM * 3 * H_DIM * 2);
  bh16*  phB    = (bh16*)alloc(slice * 2);
  float* pcF    = (float*)alloc(slice * 4);
  float* gatesF = (float*)alloc((size_t)B_DIM * 4 * H_DIM * 4);
  bh16*  finB   = (bh16*)alloc((size_t)B_DIM * 2 * H_DIM * 2);
  float* hidF   = (float*)alloc((size_t)B_DIM * FC_HID * 4);
  float* rawF   = (float*)alloc((size_t)B_DIM * 4);

  // one-pass weight conversion to bf16
  cvt_f32_bf16<<<2048, 256, 0, stream>>>(Wh,   wWh,  NCELL * H_DIM * 3 * H_DIM);
  cvt_f32_bf16<<<2048, 256, 0, stream>>>(Wc,   wWc,  NCELL * H_DIM * 3 * H_DIM);
  cvt_f32_bf16<<<2048, 256, 0, stream>>>(W_hh, wWhh, NCELL * 4 * H_DIM * H_DIM);
  cvt_f32_bf16<<<256,  256, 0, stream>>>(W1,   wW1,  FC_HID * 2 * H_DIM);

  const size_t cellW = (size_t)H_DIM * 3 * H_DIM;
  for (int i = 0; i < GHH; ++i) {
    for (int j = 0; j < GWW; ++j) {
      int k = i * GWW + j;
      const float *hL = nullptr, *cL = nullptr, *hU = nullptr, *cU = nullptr;
      if (i == 0 && j == 0)      { hL = h_ext; cL = c_ext; }
      else if (j > 0)            { hL = new_gh + (size_t)(k - 1) * slice;
                                   cL = new_gc + (size_t)(k - 1) * slice; }
      if (i > 0)                 { hU = new_gh + (size_t)(k - GWW) * slice;
                                   cU = new_gc + (size_t)(k - GWW) * slice; }
      const float* hP = grid_h + (size_t)k * slice;
      const float* cP = grid_c + (size_t)k * slice;

      prep_cell<<<2048, 256, 0, stream>>>(hL, hU, hP, cL, cU, cP, combH, combC);

      dim3 gPH(H_DIM / 128, B_DIM / 128);
      gemm_bf16<0><<<gPH, 256, 0, stream>>>(combH, wWh + (size_t)k * cellW, bh + k * H_DIM,
                                            nullptr, nullptr, nullptr, 0,
                                            3 * H_DIM, H_DIM, nullptr, phB);
      gemm_bf16<1><<<gPH, 256, 0, stream>>>(combC, wWc + (size_t)k * cellW, bc + k * H_DIM,
                                            nullptr, nullptr, nullptr, 0,
                                            3 * H_DIM, H_DIM, pcF, nullptr);
      dim3 gG(4 * H_DIM / 128, B_DIM / 128);
      gemm_bf16<2><<<gG, 256, 0, stream>>>(phB, wWhh + (size_t)k * 4 * H_DIM * H_DIM,
                                           b_ih + k * 4 * H_DIM, b_hh + k * 4 * H_DIM,
                                           x, W_ih + k * 4 * H_DIM, k,
                                           H_DIM, 4 * H_DIM, gatesF, nullptr);
      cell_update<<<2048, 256, 0, stream>>>(gatesF, pcF,
                                            new_gh + (size_t)k * slice,
                                            new_gc + (size_t)k * slice);
    }
  }

  final_prep<<<2048, 256, 0, stream>>>(new_gh + (size_t)(NCELL - 1) * slice,
                                       new_gc + (size_t)(NCELL - 1) * slice,
                                       h_ext, c_ext, final_h, final_c, finB);
  dim3 gF(FC_HID / 128, B_DIM / 128);
  gemm_bf16<3><<<gF, 256, 0, stream>>>(finB, wW1, b1, nullptr, nullptr, nullptr, 0,
                                       2 * H_DIM, FC_HID, hidF, nullptr);
  fc2_dot<<<B_DIM / 8, 256, 0, stream>>>(hidF, W2, b2, rawF);
  bn_sigmoid<<<1, 1024, 0, stream>>>(rawF, gamma, beta, out);
}
